// Gemma2Attention_59554016526977
// MI455X (gfx1250) — matrix operand order repaired and checked by reading, not compiled, not run on hardware
//
#include <hip/hip_runtime.h>

#define T_LEN 3072
#define H 16
#define KV 8
#define D 256
#define HID 3584
#define QKV_N 8192
#define WINDOW 1535
#define SCALING 0.0625f
#define SOFTCAP 50.0f
#define NEG_INF (-3.0e38f)

typedef __attribute__((ext_vector_type(4)))  unsigned int u32x4;
typedef __attribute__((ext_vector_type(8)))  int          i32x8;
typedef __attribute__((ext_vector_type(4)))  int          i32x4;
typedef __attribute__((ext_vector_type(16))) __bf16       bf16x16;
typedef __attribute__((ext_vector_type(8)))  float        f32x8;

union Frag { u32x4 u[2]; bf16x16 v; };
static_assert(sizeof(bf16x16) == 32, "bf16x16 size");
static_assert(sizeof(Frag) == 32, "frag size");

// Tensor Data Mover availability (device pass only; host pass -> 0)
#if defined(__AMDGCN__) && defined(__has_builtin)
#  if __has_builtin(__builtin_amdgcn_tensor_load_to_lds) && \
      __has_builtin(__builtin_amdgcn_s_wait_tensorcnt)
#    define HAVE_TDM 1
#  else
#    define HAVE_TDM 0
#  endif
#else
#  define HAVE_TDM 0
#endif

#if HAVE_TDM
// ---------------------------------------------------------------------------
// Issue a TDM 2D tile load: tile_d0 contiguous elements (bf16) per row,
// tile_d1 rows, row stride `stride_el` elements. LDS layout = packed rows.
// Descriptor per CDNA5 ISA 8.3/8.4 (group0: count/lds/global/type=2,
// group1: data_size=1(2B), tensor dims == tile dims, dim0 stride).
// ---------------------------------------------------------------------------
__device__ __forceinline__ void tdm_load_2d_bf16(unsigned lds_addr,
                                                 const void* gptr,
                                                 unsigned tile_d0,
                                                 unsigned tile_d1,
                                                 unsigned stride_el) {
    unsigned long long ga = (unsigned long long)(uintptr_t)gptr;
    unsigned lds_u = (unsigned)__builtin_amdgcn_readfirstlane((int)lds_addr);
    unsigned galo  = (unsigned)__builtin_amdgcn_readfirstlane((int)(unsigned)ga);
    unsigned gahi  = (unsigned)__builtin_amdgcn_readfirstlane((int)(unsigned)(ga >> 32));
    unsigned strd  = (unsigned)__builtin_amdgcn_readfirstlane((int)stride_el);

    u32x4 g0;
    g0[0] = 1u;                        // count=1, is_restore=0
    g0[1] = lds_u;                     // lds_addr (bytes)
    g0[2] = galo;                      // global_addr[31:0]
    g0[3] = gahi | (2u << 30);         // global_addr[56:32] | type=2 ("image")

    i32x8 g1;
    g1[0] = (int)(1u << 16);                                    // wg_mask=0, data_size=1 (2B)
    g1[1] = (int)((tile_d0 & 0xffffu) << 16);                   // tensor_dim0[15:0]
    g1[2] = (int)(((tile_d0 >> 16) & 0xffffu) |
                  ((tile_d1 & 0xffffu) << 16));                 // tdim0 hi | tdim1 lo
    g1[3] = (int)(((tile_d1 >> 16) & 0xffffu) |
                  ((tile_d0 & 0xffffu) << 16));                 // tdim1 hi | tile_dim0
    g1[4] = (int)(tile_d1 & 0xffffu);                           // tile_dim1 | tile_dim2=0
    g1[5] = (int)strd;                                          // dim0 stride [31:0]
    g1[6] = 0;                                                  // stride hi | dim1 stride lo
    g1[7] = 0;

    i32x4 z4 = {0, 0, 0, 0};
#if __clang_major__ >= 23
    i32x8 z8 = {0, 0, 0, 0, 0, 0, 0, 0};
    __builtin_amdgcn_tensor_load_to_lds(g0, g1, z4, z4, z8, 0);
#else
    __builtin_amdgcn_tensor_load_to_lds(g0, g1, z4, z4, 0);
#endif
}
#endif  // HAVE_TDM

// ---------------------------------------------------------------------------
// fp32 -> bf16 conversion (grid-stride)
// ---------------------------------------------------------------------------
__global__ void cvt_f32_bf16(const float* __restrict__ src,
                             __bf16* __restrict__ dst, long n) {
    long i = (long)blockIdx.x * blockDim.x + threadIdx.x;
    long stride = (long)gridDim.x * blockDim.x;
    for (; i < n; i += stride) dst[i] = (__bf16)src[i];
}

// ---------------------------------------------------------------------------
// C[M,N] (f32) = A[M,K] (bf16, row-major) * Bt[N,K]^T (bf16, row-major)
// Block: 256 threads = 8 waves. Block tile M=128 (16 rows/wave), N=64.
// B tile (64x32 bf16) staged via TDM, double-buffered in LDS.
// ---------------------------------------------------------------------------
__global__ __launch_bounds__(256) void gemm_bf16_wmma(
    const __bf16* __restrict__ A, const __bf16* __restrict__ Bt,
    float* __restrict__ C, int M, int N, int K) {
    __shared__ __bf16 ldsB[2][64 * 32];  // [buf][n][k]

    const int lane    = threadIdx.x & 31;
    const int wave    = threadIdx.x >> 5;
    const int lm      = lane & 15;
    const int halfSel = (lane >> 4) & 1;  // 0: lanes 0-15, 1: lanes 16-31
    const int rowBase = blockIdx.x * 128 + wave * 16;
    const int colBase = blockIdx.y * 64;
    const int steps   = K / 32;

    f32x8 acc[4] = {};
    const __bf16* aRow = A + (long)(rowBase + lm) * K;

    // --- stage one 64x32 Bt tile into ldsB[buf] -----------------------------
    auto stage = [&](int buf, int k0) {
#if HAVE_TDM
        if (threadIdx.x < 32) {  // one TDM issue per block (wave 0)
            tdm_load_2d_bf16((unsigned)(uintptr_t)&ldsB[buf][0],
                             Bt + (long)colBase * K + k0,
                             /*tile_d0=*/32, /*tile_d1=*/64,
                             /*stride_el=*/(unsigned)K);
        }
#else
        int t  = threadIdx.x;
        int n  = t >> 2;
        int kk = (t & 3) * 8;
        *(u32x4*)(&ldsB[buf][n * 32 + kk]) =
            *(const u32x4*)(Bt + (long)(colBase + n) * K + k0 + kk);
#endif
    };

    stage(0, 0);  // prologue: tile 0 in flight

    for (int i = 0; i < steps; ++i) {
        const int k0  = i * 32;
        const int buf = i & 1;

        if (i + 1 < steps) {
            stage(buf ^ 1, k0 + 32);  // prefetch next tile into other buffer
#if HAVE_TDM
            if (threadIdx.x < 32) __builtin_amdgcn_s_wait_tensorcnt(1);
#endif
        } else {
#if HAVE_TDM
            if (threadIdx.x < 32) __builtin_amdgcn_s_wait_tensorcnt(0);
#endif
        }
        __syncthreads();  // tile i visible to all waves

        // A fragment (16x32): lanes 0-15 K={0..7,16..23}, 16-31 K={8..15,24..31}
        Frag a;
        {
            const __bf16* ap = aRow + k0 + halfSel * 8;
            a.u[0] = *(const u32x4*)(ap);
            a.u[1] = *(const u32x4*)(ap + 16);
            __builtin_prefetch(ap + 64, 0, 3);  // global_prefetch next A tile
        }

        // 4 B fragments (32x16 each) from LDS + 4 WMMAs
        #pragma unroll
        for (int nf = 0; nf < 4; ++nf) {
            Frag b;
            const __bf16* bp = &ldsB[buf][(nf * 16 + lm) * 32 + halfSel * 8];
            b.u[0] = *(const u32x4*)(bp);
            b.u[1] = *(const u32x4*)(bp + 16);
            acc[nf] = __builtin_amdgcn_wmma_f32_16x16x32_bf16(
                false, a.v, false, b.v, (short)0, acc[nf], false, false);
        }
        __syncthreads();  // reads of tile i done before it is overwritten
    }

    // C layout: VGPR r -> row = r + 8*halfSel, col = lane%16
    #pragma unroll
    for (int nf = 0; nf < 4; ++nf) {
        #pragma unroll
        for (int r = 0; r < 8; ++r) {
            int row = rowBase + r + halfSel * 8;
            int col = colBase + nf * 16 + lm;
            C[(long)row * N + col] = acc[nf][r];
        }
    }
}

// ---------------------------------------------------------------------------
// NeoX RoPE on q/k from fp32 qkv, split to bf16 q/k/v. SCALING folded into q.
// ---------------------------------------------------------------------------
__global__ __launch_bounds__(256) void rope_split_kernel(
    const float* __restrict__ qkv, const int* __restrict__ positions,
    __bf16* __restrict__ qb, __bf16* __restrict__ kb, __bf16* __restrict__ vb) {
    const int t = blockIdx.x;
    const float pos = (float)positions[t];
    const float* row = qkv + (long)t * QKV_N;

    for (int idx = threadIdx.x; idx < (H + KV) * 128; idx += blockDim.x) {
        int head = idx >> 7;
        int d    = idx & 127;
        float inv = __powf(10000.0f, -((float)(2 * d)) * (1.0f / 256.0f));
        float fr  = pos * inv;
        float s, c;
        __sincosf(fr, &s, &c);
        const float* src;
        __bf16* dst;
        float scale;
        if (head < H) {
            src = row + head * D;
            dst = qb + ((long)t * H + head) * D;
            scale = SCALING;
        } else {
            src = row + H * D + (head - H) * D;
            dst = kb + ((long)t * KV + (head - H)) * D;
            scale = 1.0f;
        }
        float x1 = src[d], x2 = src[d + 128];
        dst[d]       = (__bf16)((x1 * c - x2 * s) * scale);
        dst[d + 128] = (__bf16)((x2 * c + x1 * s) * scale);
    }
    for (int idx = threadIdx.x; idx < KV * D; idx += blockDim.x)
        vb[(long)t * KV * D + idx] = (__bf16)row[(H + KV) * D + idx];
}

// ---------------------------------------------------------------------------
// Flash attention, sliding window + tanh softcap.
// Block: 64 threads = 2 waves; each wave owns 16 query rows of one head.
// V tile (32 keys x 256) fetched by per-wave TDM, overlapped with Q.K^T.
// ---------------------------------------------------------------------------
__global__ __launch_bounds__(64) void attn_kernel(
    const __bf16* __restrict__ qb, const __bf16* __restrict__ kb,
    const __bf16* __restrict__ vb, __bf16* __restrict__ ob) {
    __shared__ __bf16 ldsP[2][16 * 32];   // per-wave P tile  (16 q x 32 keys)
    __shared__ __bf16 ldsV[2][32 * 256];  // per-wave V tile  (32 keys x 256)

    const int lane    = threadIdx.x & 31;
    const int wave    = threadIdx.x >> 5;
    const int lm      = lane & 15;
    const int halfSel = (lane >> 4) & 1;
    const int h       = blockIdx.y;
    const int kvh     = h >> 1;  // H/KV == 2
    const int qBase   = blockIdx.x * 32 + wave * 16;

    // Q fragments: 8 chunks of 16x32 over head-dim 256 (pre-scaled by 1/16)
    bf16x16 qf[8];
    {
        const __bf16* qp = qb + ((long)(qBase + lm) * H + h) * D;
        #pragma unroll
        for (int c = 0; c < 8; ++c) {
            Frag f;
            f.u[0] = *(const u32x4*)(qp + c * 32 + halfSel * 8);
            f.u[1] = *(const u32x4*)(qp + c * 32 + halfSel * 8 + 16);
            qf[c] = f.v;
        }
    }

    f32x8 out[16] = {};
    float mrow[8], lrow[8];
    #pragma unroll
    for (int r = 0; r < 8; ++r) { mrow[r] = NEG_INF; lrow[r] = 0.0f; }

    int sLo = qBase - WINDOW;
    if (sLo < 0) sLo = 0;
    sLo &= ~31;
    const int sHi = qBase + 16;

    for (int s0 = sLo; s0 < sHi; s0 += 32) {
        // Previous iteration's LDS gathers must finish before DMA overwrite.
        asm volatile("s_wait_dscnt 0" ::: "memory");

        // Kick off V tile fetch (async; overlapped with Q.K^T below)
#if HAVE_TDM
        tdm_load_2d_bf16((unsigned)(uintptr_t)&ldsV[wave][0],
                         vb + ((long)s0 * KV + kvh) * D,
                         /*tile_d0=*/256, /*tile_d1=*/32,
                         /*stride_el=*/KV * D);
#else
        {
            const __bf16* vp = vb + ((long)(s0 + lane) * KV + kvh) * D;
            u32x4* dst = (u32x4*)&ldsV[wave][lane * 256];
            #pragma unroll
            for (int i = 0; i < 16; ++i) dst[i] = ((const u32x4*)vp)[i];
        }
#endif

        // Scores for two 16-key subtiles
        f32x8 sc[2] = {};
        #pragma unroll
        for (int t = 0; t < 2; ++t) {
            const int key = s0 + t * 16 + lm;
            const __bf16* kp = kb + ((long)key * KV + kvh) * D;
            #pragma unroll
            for (int c = 0; c < 8; ++c) {
                Frag kf;  // B frag 32x16: lane=col(key), K rows {8*halfSel..+7, 16+8*halfSel..+7}
                kf.u[0] = *(const u32x4*)(kp + c * 32 + halfSel * 8);
                kf.u[1] = *(const u32x4*)(kp + c * 32 + halfSel * 8 + 16);
                sc[t] = __builtin_amdgcn_wmma_f32_16x16x32_bf16(
                    false, qf[c], false, kf.v, (short)0, sc[t], false, false);
            }
        }

        // Softcap + mask + online softmax over the 32-key block
        float alpha[8];
        #pragma unroll
        for (int r = 0; r < 8; ++r) {
            const int i  = qBase + r + halfSel * 8;
            const int j0 = s0 + lm;
            const int j1 = s0 + 16 + lm;

            float v0 = SOFTCAP * tanhf(sc[0][r] * (1.0f / SOFTCAP));
            float v1 = SOFTCAP * tanhf(sc[1][r] * (1.0f / SOFTCAP));
            v0 = ((j0 <= i) && (i - j0 <= WINDOW)) ? v0 : NEG_INF;
            v1 = ((j1 <= i) && (i - j1 <= WINDOW)) ? v1 : NEG_INF;

            float nm = fmaxf(v0, v1);
            nm = fmaxf(nm, __shfl_xor(nm, 1, 32));
            nm = fmaxf(nm, __shfl_xor(nm, 2, 32));
            nm = fmaxf(nm, __shfl_xor(nm, 4, 32));
            nm = fmaxf(nm, __shfl_xor(nm, 8, 32));
            float mnew = fmaxf(mrow[r], nm);

            float p0, p1, a;
            if (mnew < -1e37f) {  // fully-masked block
                p0 = 0.0f; p1 = 0.0f; a = 1.0f;
            } else {
                p0 = __expf(v0 - mnew);
                p1 = __expf(v1 - mnew);
                a  = (mrow[r] < -1e37f) ? 0.0f : __expf(mrow[r] - mnew);
            }
            float ps = p0 + p1;
            ps += __shfl_xor(ps, 1, 32);
            ps += __shfl_xor(ps, 2, 32);
            ps += __shfl_xor(ps, 4, 32);
            ps += __shfl_xor(ps, 8, 32);
            lrow[r] = lrow[r] * a + ps;
            mrow[r] = mnew;
            alpha[r] = a;

            const int prow = r + halfSel * 8;
            ldsP[wave][prow * 32 + lm]      = (__bf16)p0;
            ldsP[wave][prow * 32 + 16 + lm] = (__bf16)p1;
        }

        // Rescale running output accumulators
        #pragma unroll
        for (int f = 0; f < 16; ++f)
            #pragma unroll
            for (int r = 0; r < 8; ++r) out[f][r] *= alpha[r];

#if HAVE_TDM
        __builtin_amdgcn_s_wait_tensorcnt(0);   // V tile landed in LDS
#endif
        asm volatile("s_wait_dscnt 0" ::: "memory");  // P writes visible

        // P as A fragment (16 q x 32 keys)
        Frag pf;
        {
            const __bf16* pp = &ldsP[wave][lm * 32 + halfSel * 8];
            pf.u[0] = *(const u32x4*)(pp);
            pf.u[1] = *(const u32x4*)(pp + 16);
        }

        // P.V : 16 WMMAs across the 256-wide head dim
        #pragma unroll
        for (int f = 0; f < 16; ++f) {
            bf16x16 vv;
            const int col = f * 16 + lm;
            #pragma unroll
            for (int kk = 0; kk < 16; ++kk)
                vv[kk] = ldsV[wave][(halfSel * 8 + (kk & 7) + (kk >> 3) * 16) * 256 + col];
            out[f] = __builtin_amdgcn_wmma_f32_16x16x32_bf16(
                false, pf.v, false, vv, (short)0, out[f], false, false);
        }
    }

    // Normalize and store bf16 attention output (T, H, D)
    #pragma unroll
    for (int f = 0; f < 16; ++f) {
        #pragma unroll
        for (int r = 0; r < 8; ++r) {
            const int row = qBase + r + halfSel * 8;
            float val = out[f][r] / lrow[r];
            ob[((long)row * H + h) * D + f * 16 + lm] = (__bf16)val;
        }
    }
}

// ---------------------------------------------------------------------------
extern "C" void kernel_launch(void* const* d_in, const int* in_sizes, int n_in,
                              void* d_out, int out_size, void* d_ws, size_t ws_size,
                              hipStream_t stream) {
    const int*   positions = (const int*)d_in[0];
    const float* hidden    = (const float*)d_in[1];
    const float* w_qkv     = (const float*)d_in[2];
    const float* w_o       = (const float*)d_in[3];
    float*       out       = (float*)d_out;

    // Workspace layout (bf16 copies + fp32 qkv intermediate)
    __bf16* hb  = (__bf16*)d_ws;                       // T x HID
    __bf16* wqb = hb  + (size_t)T_LEN * HID;           // QKV_N x HID
    __bf16* wob = wqb + (size_t)QKV_N * HID;           // HID x (H*D)
    __bf16* qb  = wob + (size_t)HID * (H * D);         // T x H x D (post-RoPE, pre-scaled)
    __bf16* kbb = qb  + (size_t)T_LEN * H * D;         // T x KV x D (post-RoPE)
    __bf16* vbb = kbb + (size_t)T_LEN * KV * D;        // T x KV x D
    __bf16* ab  = vbb + (size_t)T_LEN * KV * D;        // T x H x D (attention out)
    float*  qkv = (float*)(ab + (size_t)T_LEN * H * D);// T x QKV_N fp32

    cvt_f32_bf16<<<2048, 256, 0, stream>>>(hidden, hb,  (long)T_LEN * HID);
    cvt_f32_bf16<<<2048, 256, 0, stream>>>(w_qkv,  wqb, (long)QKV_N * HID);
    cvt_f32_bf16<<<2048, 256, 0, stream>>>(w_o,    wob, (long)HID * H * D);

    gemm_bf16_wmma<<<dim3(T_LEN / 128, QKV_N / 64), 256, 0, stream>>>(
        hb, wqb, qkv, T_LEN, QKV_N, HID);

    rope_split_kernel<<<T_LEN, 256, 0, stream>>>(qkv, positions, qb, kbb, vbb);

    attn_kernel<<<dim3(T_LEN / 32, H), 64, 0, stream>>>(qb, kbb, vbb, ab);

    gemm_bf16_wmma<<<dim3(T_LEN / 128, HID / 64), 256, 0, stream>>>(
        ab, wob, out, T_LEN, HID, H * D);
}
